// custom_network_11_81664508166381
// MI455X (gfx1250) — matrix operand order repaired and checked by reading, not compiled, not run on hardware
//
#include <hip/hip_runtime.h>

// ---------------- CDNA5 WMMA types ----------------
typedef __attribute__((ext_vector_type(16))) __bf16        bf16x16;
typedef __attribute__((ext_vector_type(8)))  float         f32x8;
typedef __attribute__((ext_vector_type(4)))  unsigned int  u32x4;

union Frag {
  u32x4   u[2];   // two 16-byte halves
  bf16x16 v;      // 16 bf16 lanes-worth of fragment data
};

static __device__ __forceinline__ f32x8 wmma_bf16(const Frag& a, const Frag& b, f32x8 c) {
  // D = A(16x32 bf16) * B(32x16 bf16) + C(16x16 f32)
  return __builtin_amdgcn_wmma_f32_16x16x32_bf16(false, a.v, false, b.v, (short)0, c, false, false);
}

// ---------------- Layer 1: naive fp32 conv (Ci=3, ~1% of FLOPs) ----------------
// x: [64,3,64,64] NCHW fp32 ; w: [64,3,3,3] ; out: NHWC fp32 [64,64,64,64] as [m][64]
__global__ void conv1_naive(const float* __restrict__ x, const float* __restrict__ w,
                            float* __restrict__ convOut) {
  long long idx = (long long)blockIdx.x * blockDim.x + threadIdx.x;
  const long long total = 64LL * 64 * 64 * 64;
  if (idx >= total) return;
  int co = (int)(idx & 63);
  int m  = (int)(idx >> 6);
  int n  = m >> 12;
  int r  = m & 4095;
  int ho = r >> 6;
  int wo = r & 63;
  float s = 0.f;
  for (int ci = 0; ci < 3; ++ci)
    for (int kh = 0; kh < 3; ++kh) {
      int hi = ho + kh - 1; if (hi < 0 || hi >= 64) continue;
      for (int kw = 0; kw < 3; ++kw) {
        int wi = wo + kw - 1; if (wi < 0 || wi >= 64) continue;
        s += x[((n * 3 + ci) * 64 + hi) * 64 + wi] * w[((co * 3 + ci) * 3 + kh) * 3 + kw];
      }
    }
  convOut[idx] = s;
}

// ---------------- Implicit-GEMM conv via WMMA (layers 2..9) ----------------
// act: NHWC bf16 [64,Hi,Wi,Ci]; wpack: bf16 [Co][9*Ci] (K = tap*Ci+ci)
// convOut: fp32 [M][Co]. Wave macro-tile: (16*MBLK) x (16*NBLK); block = 4 waves along M.
template <int MBLK, int NBLK>
__global__ void __launch_bounds__(128)
conv_wmma(const __bf16* __restrict__ act, const __bf16* __restrict__ wpack,
          float* __restrict__ convOut,
          int Hi, int Wi, int Ci, int Ho, int Wo, int Co, int stride) {
  const int lane  = threadIdx.x & 31;
  const int wave  = threadIdx.x >> 5;
  const int mBase = (blockIdx.x * 4 + wave) * (16 * MBLK);
  const int nBase = blockIdx.y * (16 * NBLK);
  const int HoWo  = Ho * Wo;
  const int Mtot  = 64 * HoWo;
  if (mBase >= Mtot) return;

  int n_[MBLK], ho_[MBLK], wo_[MBLK];
  #pragma unroll
  for (int mb = 0; mb < MBLK; ++mb) {
    int m = mBase + mb * 16 + (lane & 15);      // lanes 0-15 / 16-31 share M rows
    int n = m / HoWo;
    int r = m - n * HoWo;
    int h = r / Wo;
    n_[mb] = n; ho_[mb] = h; wo_[mb] = r - h * Wo;
  }
  const int kbA  = (lane >> 4) * 8;             // A: K sub-base per half-wave
  const int Ktot = 9 * Ci;
  const __bf16* brow[NBLK];
  #pragma unroll
  for (int t = 0; t < NBLK; ++t)
    brow[t] = wpack + (size_t)(nBase + t * 16 + (lane & 15)) * Ktot + (lane >> 4) * 8;

  f32x8 acc[MBLK][NBLK];
  const f32x8 zf = {0.f, 0.f, 0.f, 0.f, 0.f, 0.f, 0.f, 0.f};
  #pragma unroll
  for (int mb = 0; mb < MBLK; ++mb)
    #pragma unroll
    for (int t = 0; t < NBLK; ++t) acc[mb][t] = zf;
  const u32x4 zero = {0u, 0u, 0u, 0u};

  for (int tap = 0; tap < 9; ++tap) {
    const int kh = tap / 3, kw = tap - kh * 3;
    const __bf16* abase[MBLK];
    bool ok[MBLK];
    #pragma unroll
    for (int mb = 0; mb < MBLK; ++mb) {
      int hi = ho_[mb] * stride + kh - 1;
      int wi = wo_[mb] * stride + kw - 1;
      ok[mb] = (hi >= 0) && (hi < Hi) && (wi >= 0) && (wi < Wi);
      abase[mb] = act + ((size_t)(n_[mb] * Hi + hi) * Wi + wi) * Ci + kbA;
    }
    for (int c0 = 0; c0 < Ci; c0 += 32) {       // 32-wide K chunk stays inside one tap
      Frag aF[MBLK];
      #pragma unroll
      for (int mb = 0; mb < MBLK; ++mb) {
        if (ok[mb]) {
          aF[mb].u[0] = *(const u32x4*)(abase[mb] + c0);
          aF[mb].u[1] = *(const u32x4*)(abase[mb] + c0 + 16);
        } else {
          aF[mb].u[0] = zero; aF[mb].u[1] = zero;
        }
      }
      __builtin_prefetch(brow[0] + tap * Ci + c0 + 32, 0, 0);   // global_prefetch_b8
      #pragma unroll
      for (int t = 0; t < NBLK; ++t) {
        Frag bF;
        bF.u[0] = *(const u32x4*)(brow[t] + tap * Ci + c0);
        bF.u[1] = *(const u32x4*)(brow[t] + tap * Ci + c0 + 16);
        #pragma unroll
        for (int mb = 0; mb < MBLK; ++mb)
          acc[mb][t] = wmma_bf16(aF[mb], bF, acc[mb][t]);
      }
    }
  }
  #pragma unroll
  for (int mb = 0; mb < MBLK; ++mb)
    #pragma unroll
    for (int t = 0; t < NBLK; ++t) {
      int co = nBase + t * 16 + (lane & 15);
      #pragma unroll
      for (int rr = 0; rr < 8; ++rr) {          // D: VGPR rr -> M = rr + (lane>>4)*8
        int mr = mBase + mb * 16 + (lane >> 4) * 8 + rr;
        convOut[(size_t)mr * Co + co] = acc[mb][t][rr];
      }
    }
}

// ---------------- Generic WMMA GEMM (MLP layers) ----------------
// A: bf16 [M][K]; Bt: bf16 [Npad][K] (pre-transposed weights); out row stride = Ntrue
template <int MBLK, int NBLK>
__global__ void __launch_bounds__(128)
gemm_wmma(const __bf16* __restrict__ A, const __bf16* __restrict__ Bt,
          const float* __restrict__ bias,
          __bf16* __restrict__ Cb, float* __restrict__ Cf,
          int M, int Ntrue, int K, int relu) {
  const int lane  = threadIdx.x & 31;
  const int wave  = threadIdx.x >> 5;
  const int mBase = (blockIdx.x * 4 + wave) * (16 * MBLK);
  const int nBase = blockIdx.y * (16 * NBLK);
  if (mBase >= M) return;

  const __bf16* arow[MBLK];
  #pragma unroll
  for (int mb = 0; mb < MBLK; ++mb)
    arow[mb] = A + (size_t)(mBase + mb * 16 + (lane & 15)) * K + (lane >> 4) * 8;
  const __bf16* brow[NBLK];
  #pragma unroll
  for (int t = 0; t < NBLK; ++t)
    brow[t] = Bt + (size_t)(nBase + t * 16 + (lane & 15)) * K + (lane >> 4) * 8;

  f32x8 acc[MBLK][NBLK];
  const f32x8 zf = {0.f, 0.f, 0.f, 0.f, 0.f, 0.f, 0.f, 0.f};
  #pragma unroll
  for (int mb = 0; mb < MBLK; ++mb)
    #pragma unroll
    for (int t = 0; t < NBLK; ++t) acc[mb][t] = zf;

  for (int k0 = 0; k0 < K; k0 += 32) {
    Frag aF[MBLK];
    #pragma unroll
    for (int mb = 0; mb < MBLK; ++mb) {
      aF[mb].u[0] = *(const u32x4*)(arow[mb] + k0);
      aF[mb].u[1] = *(const u32x4*)(arow[mb] + k0 + 16);
    }
    __builtin_prefetch(arow[0] + k0 + 64, 0, 0);                // global_prefetch_b8
    #pragma unroll
    for (int t = 0; t < NBLK; ++t) {
      Frag bF;
      bF.u[0] = *(const u32x4*)(brow[t] + k0);
      bF.u[1] = *(const u32x4*)(brow[t] + k0 + 16);
      #pragma unroll
      for (int mb = 0; mb < MBLK; ++mb)
        acc[mb][t] = wmma_bf16(aF[mb], bF, acc[mb][t]);
    }
  }
  #pragma unroll
  for (int t = 0; t < NBLK; ++t) {
    const int ncol = nBase + t * 16 + (lane & 15);
    if (ncol >= Ntrue) continue;
    const float bv = bias ? bias[ncol] : 0.f;
    #pragma unroll
    for (int mb = 0; mb < MBLK; ++mb)
      #pragma unroll
      for (int rr = 0; rr < 8; ++rr) {
        int mr = mBase + mb * 16 + (lane >> 4) * 8 + rr;
        float v = acc[mb][t][rr] + bv;
        if (relu) v = fmaxf(v, 0.f);
        if (Cb) Cb[(size_t)mr * Ntrue + ncol] = (__bf16)v;
        if (Cf) Cf[(size_t)mr * Ntrue + ncol] = v;
      }
  }
}

// ---------------- BN statistics (per channel, double accumulation) ----------------
__global__ void __launch_bounds__(256)
bn_stats(const float* __restrict__ convOut, const float* __restrict__ g,
         const float* __restrict__ b, float* __restrict__ scale,
         float* __restrict__ shift, int M, int C) {
  const int c = blockIdx.x;
  const int tid = threadIdx.x;
  double s = 0.0, s2 = 0.0;
  for (int i = tid; i < M; i += 256) {
    float v = convOut[(size_t)i * C + c];
    s += v; s2 += (double)v * v;
  }
  __shared__ double sh1[256], sh2[256];
  sh1[tid] = s; sh2[tid] = s2;
  __syncthreads();
  for (int off = 128; off > 0; off >>= 1) {
    if (tid < off) { sh1[tid] += sh1[tid + off]; sh2[tid] += sh2[tid + off]; }
    __syncthreads();
  }
  if (tid == 0) {
    double mean = sh1[0] / M;
    double var  = sh2[0] / M - mean * mean;
    float inv = (float)(1.0 / sqrt(var + 1e-5));
    float sc  = g[c] * inv;
    scale[c] = sc;
    shift[c] = b[c] - (float)mean * sc;
  }
}

// ---------------- Fused BN + LeakyReLU(0.2) + bf16 cast ----------------
__global__ void bn_apply(const float* __restrict__ convOut, const float* __restrict__ scale,
                         const float* __restrict__ shift, __bf16* __restrict__ act,
                         long long total, int C) {
  long long idx = (long long)blockIdx.x * blockDim.x + threadIdx.x;
  if (idx >= total) return;
  int c = (int)(idx % C);
  float v = convOut[idx] * scale[c] + shift[c];
  v = v > 0.f ? v : 0.2f * v;
  act[idx] = (__bf16)v;
}

// ---------------- Weight packing ----------------
// conv: OIHW fp32 -> [Co][tap][Ci] bf16
__global__ void pack_conv(const float* __restrict__ w, __bf16* __restrict__ dst,
                          int Ci, int total) {
  int idx = blockIdx.x * blockDim.x + threadIdx.x;
  if (idx >= total) return;
  int co  = idx / (9 * Ci);
  int r   = idx - co * 9 * Ci;
  int tap = r / Ci;
  int ci  = r - tap * Ci;
  dst[idx] = (__bf16)w[(co * Ci + ci) * 9 + tap];
}

// linear: [K][N] fp32 -> [Npad][K] bf16, zero-padded rows
__global__ void pack_lin(const float* __restrict__ w, __bf16* __restrict__ dst,
                         int K, int N, int Npad) {
  int idx = blockIdx.x * blockDim.x + threadIdx.x;
  if (idx >= Npad * K) return;
  int n = idx / K;
  int k = idx - n * K;
  dst[idx] = (n < N) ? (__bf16)w[(size_t)k * N + n] : (__bf16)0.f;
}

// ---------------- Pair-concat feature builders ----------------
// act9: NHWC bf16 [64,8,8,128]. cat_s: [120*64][1024] bf16
__global__ void build_cat_s(const __bf16* __restrict__ act9, __bf16* __restrict__ dst) {
  int idx = blockIdx.x * blockDim.x + threadIdx.x;
  if (idx >= 120 * 64 * 1024) return;
  int f = idx & 1023;
  int n = (idx >> 10) & 63;
  int p = idx >> 16;
  int a = 0, rem = p;
  while (rem >= 15 - a) { rem -= 15 - a; ++a; }
  int bq = a + 1 + rem;
  int q  = (f < 512) ? a : bq;
  int f2 = f & 511;
  int ch = f2 >> 2;
  int h2 = (f2 >> 1) & 1;
  int w2 = f2 & 1;
  int H = (q >> 2) * 2 + h2;
  int W = (q & 3) * 2 + w2;
  dst[idx] = act9[((n * 8 + H) * 8 + W) * 128 + ch];
}

// cat_b: [6*64][4096] bf16
__global__ void build_cat_b(const __bf16* __restrict__ act9, __bf16* __restrict__ dst) {
  int idx = blockIdx.x * blockDim.x + threadIdx.x;
  if (idx >= 6 * 64 * 4096) return;
  int f = idx & 4095;
  int n = (idx >> 12) & 63;
  int p = idx >> 18;
  int a = 0, rem = p;
  while (rem >= 3 - a) { rem -= 3 - a; ++a; }
  int bq = a + 1 + rem;
  int q  = (f < 2048) ? a : bq;
  int f2 = f & 2047;
  int ch = f2 >> 4;
  int h2 = (f2 >> 2) & 3;
  int w2 = f2 & 3;
  int H = (q >> 1) * 4 + h2;
  int W = (q & 1) * 4 + w2;
  dst[idx] = act9[((n * 8 + H) * 8 + W) * 128 + ch];
}

// ---------------- Pair sum: acc[n][c] (+)= sum_p src[(p*64+n)][c] ----------------
__global__ void pair_sum(const __bf16* __restrict__ src, float* __restrict__ acc,
                         int P, int C, int init) {
  int idx = blockIdx.x * blockDim.x + threadIdx.x;
  if (idx >= 64 * C) return;
  int n = idx / C;
  int c = idx - n * C;
  float s = init ? 0.f : acc[idx];
  for (int p = 0; p < P; ++p)
    s += (float)src[(size_t)(p * 64 + n) * C + c];
  acc[idx] = s;
}

__global__ void f32_to_bf16(const float* __restrict__ src, __bf16* __restrict__ dst, int n) {
  int idx = blockIdx.x * blockDim.x + threadIdx.x;
  if (idx < n) dst[idx] = (__bf16)src[idx];
}

// ---------------- Host orchestration ----------------
extern "C" void kernel_launch(void* const* d_in, const int* in_sizes, int n_in,
                              void* d_out, int out_size, void* d_ws, size_t ws_size,
                              hipStream_t stream) {
  (void)in_sizes; (void)n_in; (void)out_size; (void)ws_size;
  const float* x = (const float*)d_in[0];
  const float *convW[9], *bnG[9], *bnB[9];
  for (int i = 0; i < 9; ++i) {
    convW[i] = (const float*)d_in[1 + i];
    bnG[i]   = (const float*)d_in[10 + i];
    bnB[i]   = (const float*)d_in[19 + i];
  }
  const float *gsW[4], *gsB[4], *gbW[4], *gbB[4];
  for (int i = 0; i < 4; ++i) {
    gsW[i] = (const float*)d_in[28 + i];
    gsB[i] = (const float*)d_in[32 + i];
    gbW[i] = (const float*)d_in[36 + i];
    gbB[i] = (const float*)d_in[40 + i];
  }
  const float *fpW[3], *fpB[3];
  for (int i = 0; i < 3; ++i) {
    fpW[i] = (const float*)d_in[44 + i];
    fpB[i] = (const float*)d_in[47 + i];
  }
  float* out = (float*)d_out;

  char* wsp = (char*)d_ws;
  auto alloc = [&](size_t bytes) -> void* {
    void* r = (void*)wsp;
    wsp += (bytes + 255) & ~(size_t)255;
    return r;
  };

  float*  convOut = (float*) alloc((size_t)262144 * 64 * sizeof(float));  // largest conv output
  __bf16* actA    = (__bf16*)alloc((size_t)16777216 * sizeof(__bf16));    // largest NHWC act
  __bf16* actB    = (__bf16*)alloc((size_t)16777216 * sizeof(__bf16));
  float*  scaleB  = (float*) alloc(128 * sizeof(float));
  float*  shiftB  = (float*) alloc(128 * sizeof(float));

  static const int CiA[9] = {3, 64, 64, 128, 128, 128, 128, 128, 128};
  static const int CoA[9] = {64, 64, 128, 128, 128, 128, 128, 128, 128};
  static const int StA[9] = {1, 1, 2, 1, 1, 2, 1, 1, 2};
  static const int HiA[9] = {64, 64, 64, 32, 32, 32, 16, 16, 16};

  // Pack conv weights (layers 1..8) to bf16 [Co][tap][Ci]
  __bf16* wp[9] = {};
  for (int l = 1; l < 9; ++l) {
    int elems = CoA[l] * 9 * CiA[l];
    wp[l] = (__bf16*)alloc((size_t)elems * sizeof(__bf16));
    pack_conv<<<(elems + 255) / 256, 256, 0, stream>>>(convW[l], wp[l], CiA[l], elems);
  }

  // Pack linear weights to transposed bf16 [Npad][K]
  const int gsD[5] = {1024, 512, 256, 256, 256};
  const int gbD[5] = {4096, 2048, 1024, 512, 256};
  __bf16 *gsBt[4], *gbBt[4], *fpBt[3];
  for (int i = 0; i < 4; ++i) {
    int K = gsD[i], N = gsD[i + 1];
    gsBt[i] = (__bf16*)alloc((size_t)N * K * sizeof(__bf16));
    pack_lin<<<(N * K + 255) / 256, 256, 0, stream>>>(gsW[i], gsBt[i], K, N, N);
    K = gbD[i]; N = gbD[i + 1];
    gbBt[i] = (__bf16*)alloc((size_t)N * K * sizeof(__bf16));
    pack_lin<<<(N * K + 255) / 256, 256, 0, stream>>>(gbW[i], gbBt[i], K, N, N);
  }
  const int fpK[3]  = {256, 256, 256};
  const int fpN[3]  = {256, 256, 200};
  const int fpNp[3] = {256, 256, 208};   // pad last N to multiple of 16
  for (int i = 0; i < 3; ++i) {
    fpBt[i] = (__bf16*)alloc((size_t)fpNp[i] * fpK[i] * sizeof(__bf16));
    pack_lin<<<(fpNp[i] * fpK[i] + 255) / 256, 256, 0, stream>>>(fpW[i], fpBt[i], fpK[i], fpN[i], fpNp[i]);
  }

  __bf16* mlpA   = (__bf16*)alloc((size_t)7680 * 1024 * sizeof(__bf16));
  __bf16* mlpB   = (__bf16*)alloc((size_t)7680 * 512 * sizeof(__bf16));
  float*  relAcc = (float*) alloc((size_t)64 * 256 * sizeof(float));
  __bf16* relB   = (__bf16*)alloc((size_t)64 * 256 * sizeof(__bf16));
  __bf16* f1     = (__bf16*)alloc((size_t)64 * 256 * sizeof(__bf16));
  __bf16* f2     = (__bf16*)alloc((size_t)64 * 256 * sizeof(__bf16));

  // ---- conv stack ----
  const __bf16* actIn = nullptr;
  for (int l = 0; l < 9; ++l) {
    int Hi = HiA[l], Wi = HiA[l];
    int Ho = Hi / StA[l], Wo = Ho;
    int M = 64 * Ho * Wo;
    int C = CoA[l];
    if (l == 0) {
      long long tot = (long long)M * C;
      conv1_naive<<<(unsigned)((tot + 255) / 256), 256, 0, stream>>>(x, convW[0], convOut);
    } else {
      dim3 g(M / 128, C / 64);   // wave tile 32x64, 4 waves -> block tile 128x64
      conv_wmma<2, 4><<<g, 128, 0, stream>>>(actIn, wp[l], convOut, Hi, Wi, CiA[l], Ho, Wo, C, StA[l]);
    }
    bn_stats<<<C, 256, 0, stream>>>(convOut, bnG[l], bnB[l], scaleB, shiftB, M, C);
    __bf16* actOut = (l & 1) ? actB : actA;
    long long tot = (long long)M * C;
    bn_apply<<<(unsigned)((tot + 255) / 256), 256, 0, stream>>>(convOut, scaleB, shiftB, actOut, tot, C);
    actIn = actOut;
  }
  const __bf16* act9 = actIn;  // NHWC bf16 [64,8,8,128]

  auto gemm = [&](const __bf16* A, const __bf16* Bt, const float* bias,
                  __bf16* Cb, float* Cf, int M, int Npad, int Ntrue, int K, int relu) {
    if ((Npad & 63) == 0) {
      dim3 g((M + 127) / 128, Npad / 64);
      gemm_wmma<2, 4><<<g, 128, 0, stream>>>(A, Bt, bias, Cb, Cf, M, Ntrue, K, relu);
    } else {
      dim3 g((M + 63) / 64, Npad / 16);
      gemm_wmma<1, 1><<<g, 128, 0, stream>>>(A, Bt, bias, Cb, Cf, M, Ntrue, K, relu);
    }
  };

  // ---- g_theta over 120 small-patch pairs ----
  build_cat_s<<<(120 * 64 * 1024 + 255) / 256, 256, 0, stream>>>(act9, mlpA);
  gemm(mlpA, gsBt[0], gsB[0], mlpB, nullptr, 7680, 512, 512, 1024, 1);
  gemm(mlpB, gsBt[1], gsB[1], mlpA, nullptr, 7680, 256, 256, 512, 1);
  gemm(mlpA, gsBt[2], gsB[2], mlpB, nullptr, 7680, 256, 256, 256, 1);
  gemm(mlpB, gsBt[3], gsB[3], mlpA, nullptr, 7680, 256, 256, 256, 1);
  pair_sum<<<64, 256, 0, stream>>>(mlpA, relAcc, 120, 256, 1);

  // ---- g_phi over 6 big-patch pairs ----
  build_cat_b<<<(6 * 64 * 4096 + 255) / 256, 256, 0, stream>>>(act9, mlpB);
  gemm(mlpB, gbBt[0], gbB[0], mlpA, nullptr, 384, 2048, 2048, 4096, 1);
  gemm(mlpA, gbBt[1], gbB[1], mlpB, nullptr, 384, 1024, 1024, 2048, 1);
  gemm(mlpB, gbBt[2], gbB[2], mlpA, nullptr, 384, 512, 512, 1024, 1);
  gemm(mlpA, gbBt[3], gbB[3], mlpB, nullptr, 384, 256, 256, 512, 1);
  pair_sum<<<64, 256, 0, stream>>>(mlpB, relAcc, 6, 256, 0);

  // ---- f_phi ----
  f32_to_bf16<<<64, 256, 0, stream>>>(relAcc, relB, 64 * 256);
  gemm(relB, fpBt[0], fpB[0], f1, nullptr, 64, 256, 256, 256, 1);
  gemm(f1,   fpBt[1], fpB[1], f2, nullptr, 64, 256, 256, 256, 1);
  gemm(f2,   fpBt[2], fpB[2], nullptr, out, 64, 208, 200, 256, 0);  // fp32 -> d_out [64,200]
}
